// AttentionBlock_39556648796432
// MI455X (gfx1250) — matrix operand order repaired and checked by reading, not compiled, not run on hardware
//
#include <hip/hip_runtime.h>
#include <hip/hip_bf16.h>

// ---------------- problem constants ----------------
#define BATCH 4
#define CCH   512              // C
#define NG    8                // groups
#define CG    64               // C / G
#define NPIX  4096             // H*W
#define EPS   1e-5f
#define SCALE 0.044194173824159216f   // 1/sqrt(512)

typedef __attribute__((ext_vector_type(16))) __bf16 v16bf;
typedef __attribute__((ext_vector_type(8)))  __bf16 v8bf;
typedef __attribute__((ext_vector_type(8)))  float  v8f;

__device__ __forceinline__ v8f vzero8() {
    v8f r;
    #pragma unroll
    for (int i = 0; i < 8; ++i) r[i] = 0.f;
    return r;
}
__device__ __forceinline__ v16bf cat16(v8bf lo, v8bf hi) {
    v16bf r;
    #pragma unroll
    for (int i = 0; i < 8; ++i) { r[i] = lo[i]; r[i + 8] = hi[i]; }
    return r;
}
__device__ __forceinline__ v8f wmma_bf16(v16bf a, v16bf b, v8f c) {
    return __builtin_amdgcn_wmma_f32_16x16x32_bf16(
        /*neg_a=*/false, a, /*neg_b=*/false, b,
        /*c_mod=*/(short)0, c, /*reuse_a=*/false, /*reuse_b=*/false);
}

// ---------------- workspace layout (bytes) ----------------
#define HT_OFF   ((size_t)0)                      // h normalized  [B][N][C] bf16
#define Q_OFF    (HT_OFF + (size_t)16777216)      // q [B][N][C] bf16
#define K_OFF    (Q_OFF  + (size_t)16777216)      // k [B][N][C] bf16
#define V_OFF    (K_OFF  + (size_t)16777216)      // v [B][C][N] bf16 (transposed)
#define AO_OFF   (V_OFF  + (size_t)16777216)      // attn out [B][N][C] bf16
#define QW_OFF   (AO_OFF + (size_t)16777216)      // qkv_w bf16 (3C*C)
#define PW_OFF   (QW_OFF + (size_t)1572864)       // proj_w bf16 (C*C)
#define MU_OFF   (PW_OFF + (size_t)524288)        // 32 floats
#define RS_OFF   (MU_OFF + (size_t)128)           // 32 floats

// ---------------- 0: convert weights to bf16 ----------------
__global__ void cvt_weights(const float* __restrict__ qw, const float* __restrict__ pw,
                            __bf16* __restrict__ qwb, __bf16* __restrict__ pwb) {
    int idx = blockIdx.x * 256 + threadIdx.x;
    if (idx < 3 * CCH * CCH) qwb[idx] = (__bf16)qw[idx];
    if (idx < CCH * CCH)     pwb[idx] = (__bf16)pw[idx];
}

// ---------------- 1: groupnorm statistics ----------------
__global__ void gn_stats(const float* __restrict__ x, float* __restrict__ mu,
                         float* __restrict__ rs) {
    // one block per (b,g); data contiguous: CG*NPIX floats
    const float* p = x + (size_t)blockIdx.x * (CG * NPIX);
    float s = 0.f, s2 = 0.f;
    for (int i = threadIdx.x; i < CG * NPIX; i += 256) {
        float v = p[i]; s += v; s2 += v * v;
    }
    __shared__ float sh[256], sh2[256];
    sh[threadIdx.x] = s; sh2[threadIdx.x] = s2;
    __syncthreads();
    for (int off = 128; off > 0; off >>= 1) {
        if (threadIdx.x < off) {
            sh[threadIdx.x]  += sh[threadIdx.x + off];
            sh2[threadIdx.x] += sh2[threadIdx.x + off];
        }
        __syncthreads();
    }
    if (threadIdx.x == 0) {
        float inv = 1.f / (float)(CG * NPIX);
        float m   = sh[0] * inv;
        float var = sh2[0] * inv - m * m;
        mu[blockIdx.x] = m;
        rs[blockIdx.x] = rsqrtf(var + EPS);
    }
}

// ---------------- 2: groupnorm apply, write h^T [B][N][C] bf16 ----------------
__global__ void gn_apply(const float* __restrict__ x, const float* __restrict__ gw,
                         const float* __restrict__ gb, const float* __restrict__ mu,
                         const float* __restrict__ rs, __bf16* __restrict__ hT) {
    size_t idx = (size_t)blockIdx.x * 256 + threadIdx.x;   // over B*C*N
    int b = (int)(idx / ((size_t)CCH * NPIX));
    int c = (int)((idx / NPIX) % CCH);
    int n = (int)(idx % NPIX);
    int grp = b * NG + c / CG;
    float v = (x[idx] - mu[grp]) * rs[grp] * gw[c] + gb[c];
    hT[((size_t)b * NPIX + n) * CCH + c] = (__bf16)v;
}

// ---------------- 3: QKV GEMM (WMMA), scatter q/k/v ----------------
__global__ void __launch_bounds__(128) qkv_gemm(
    const __bf16* __restrict__ w, const __bf16* __restrict__ hT,
    const float* __restrict__ bias, __bf16* __restrict__ q,
    __bf16* __restrict__ k, __bf16* __restrict__ v) {
    int lane = threadIdx.x & 31;
    int wv   = blockIdx.x * 4 + (threadIdx.x >> 5);
    int b  = wv / (96 * 64);
    int r  = wv % (96 * 64);
    int o0 = (r / 64) * 16;        // 0..1520
    int n0 = (r % 64) * 64;
    int hi = lane >> 4, lr = lane & 15;
    const __bf16* hb = hT + (size_t)b * NPIX * CCH;

    v8f acc[4];
    #pragma unroll
    for (int t = 0; t < 4; ++t) acc[t] = vzero8();

    for (int k0 = 0; k0 < CCH; k0 += 32) {
        const __bf16* ap = w + (size_t)(o0 + lr) * CCH + k0 + hi * 8;
        v16bf af = cat16(*(const v8bf*)ap, *(const v8bf*)(ap + 16));
        #pragma unroll
        for (int t = 0; t < 4; ++t) {
            int n = n0 + t * 16 + lr;
            const __bf16* bp = hb + (size_t)n * CCH + k0 + hi * 8;
            v16bf bf_ = cat16(*(const v8bf*)bp, *(const v8bf*)(bp + 16));
            acc[t] = wmma_bf16(af, bf_, acc[t]);
        }
    }
    #pragma unroll
    for (int t = 0; t < 4; ++t) {
        int n = n0 + t * 16 + lr;
        #pragma unroll
        for (int vv = 0; vv < 8; ++vv) {
            int o = o0 + vv + hi * 8;
            float val = acc[t][vv] + bias[o];
            if (o < CCH)
                q[((size_t)b * NPIX + n) * CCH + o] = (__bf16)val;
            else if (o < 2 * CCH)
                k[((size_t)b * NPIX + n) * CCH + (o - CCH)] = (__bf16)val;
            else
                v[((size_t)b * CCH + (o - 2 * CCH)) * NPIX + n] = (__bf16)val;
        }
    }
}

// ---------------- 4: flash attention ----------------
// block = 16 query rows, 128 threads = 4 waves; wave w owns d-slice [128w,128w+128)
__global__ void __launch_bounds__(128) attn_flash(
    const __bf16* __restrict__ q, const __bf16* __restrict__ k,
    const __bf16* __restrict__ vT, __bf16* __restrict__ ao) {
    __shared__ float  Ssh[16][32];
    __shared__ __bf16 Psh[16][32];
    __shared__ float  mrow[16], lrow[16], arow[16];

    int tid  = threadIdx.x;
    int lane = tid & 31;
    int wv   = tid >> 5;
    int b  = blockIdx.x >> 8;
    int i0 = (blockIdx.x & 255) << 4;
    int d0 = wv * 128;
    int hi = lane >> 4, lr = lane & 15;

    const __bf16* qb = q  + (size_t)b * NPIX * CCH;
    const __bf16* kb = k  + (size_t)b * NPIX * CCH;
    const __bf16* vb = vT + (size_t)b * CCH * NPIX;

    if (tid < 16) { mrow[tid] = -3.0e38f; lrow[tid] = 0.f; }

    v16bf qf[4];
    #pragma unroll
    for (int kk = 0; kk < 4; ++kk) {
        const __bf16* p = qb + (size_t)(i0 + lr) * CCH + d0 + kk * 32 + hi * 8;
        qf[kk] = cat16(*(const v8bf*)p, *(const v8bf*)(p + 16));
    }
    v8f acc[8];
    #pragma unroll
    for (int t = 0; t < 8; ++t) acc[t] = vzero8();
    __syncthreads();

    for (int jb = 0; jb < NPIX; jb += 32) {
        // zero S
        for (int t = tid; t < 512; t += 128) ((float*)Ssh)[t] = 0.f;
        __syncthreads();

        // partial S over this wave's d-slice; reduce across waves via ds_add
        #pragma unroll
        for (int t = 0; t < 2; ++t) {
            v8f s = vzero8();
            int j0 = jb + t * 16;
            #pragma unroll
            for (int kk = 0; kk < 4; ++kk) {
                const __bf16* p = kb + (size_t)(j0 + lr) * CCH + d0 + kk * 32 + hi * 8;
                v16bf kf = cat16(*(const v8bf*)p, *(const v8bf*)(p + 16));
                s = wmma_bf16(qf[kk], kf, s);
            }
            #pragma unroll
            for (int vv = 0; vv < 8; ++vv)
                atomicAdd(&Ssh[vv + hi * 8][t * 16 + lr], s[vv] * SCALE);
        }
        __syncthreads();

        // online softmax bookkeeping (threads 0..15, one row each)
        if (tid < 16) {
            float mo = mrow[tid];
            float mx = mo;
            #pragma unroll
            for (int cc = 0; cc < 32; ++cc) mx = fmaxf(mx, Ssh[tid][cc]);
            float al = __expf(mo - mx);
            float ls = lrow[tid] * al;
            #pragma unroll
            for (int cc = 0; cc < 32; ++cc) {
                float pv = __expf(Ssh[tid][cc] - mx);
                ls += pv;
                Psh[tid][cc] = (__bf16)pv;
            }
            mrow[tid] = mx; lrow[tid] = ls; arow[tid] = al;
        }
        __syncthreads();

        // rescale running O by alpha[row]
        #pragma unroll
        for (int vv = 0; vv < 8; ++vv) {
            float al = arow[vv + hi * 8];
            #pragma unroll
            for (int t = 0; t < 8; ++t) acc[t][vv] *= al;
        }

        // rebuild P tile in WMMA A-fragment layout (row = lane&15, klo = 8*(lane>=16))
        v16bf pf;
        {
            int klo = hi * 8;
            #pragma unroll
            for (int e = 0; e < 8; ++e) {
                pf[e]     = Psh[lr][klo + e];
                pf[e + 8] = Psh[lr][16 + klo + e];
            }
        }
        // O += P x V  (8 output tiles across this wave's 128 columns)
        #pragma unroll
        for (int t = 0; t < 8; ++t) {
            int c0 = d0 + t * 16;
            const __bf16* p = vb + (size_t)(c0 + lr) * NPIX + jb + hi * 8;
            v16bf vf = cat16(*(const v8bf*)p, *(const v8bf*)(p + 16));
            acc[t] = wmma_bf16(pf, vf, acc[t]);
        }
        __syncthreads();
    }

    // normalize by l and store [B][N][C] bf16
    #pragma unroll
    for (int vv = 0; vv < 8; ++vv) {
        float inv = 1.f / lrow[vv + hi * 8];
        int row = i0 + vv + hi * 8;
        #pragma unroll
        for (int t = 0; t < 8; ++t) {
            int c = d0 + t * 16 + lr;
            ao[((size_t)b * NPIX + row) * CCH + c] = (__bf16)(acc[t][vv] * inv);
        }
    }
}

// ---------------- 5: proj GEMM + bias + residual (f32 out) ----------------
__global__ void __launch_bounds__(128) proj_gemm(
    const __bf16* __restrict__ w, const __bf16* __restrict__ ao,
    const float* __restrict__ bias, const float* __restrict__ x,
    float* __restrict__ out) {
    int lane = threadIdx.x & 31;
    int wv   = blockIdx.x * 4 + (threadIdx.x >> 5);
    int b  = wv / (32 * 64);
    int r  = wv % (32 * 64);
    int o0 = (r / 64) * 16;
    int n0 = (r % 64) * 64;
    int hi = lane >> 4, lr = lane & 15;
    const __bf16* ab = ao + (size_t)b * NPIX * CCH;

    v8f acc[4];
    #pragma unroll
    for (int t = 0; t < 4; ++t) acc[t] = vzero8();

    for (int k0 = 0; k0 < CCH; k0 += 32) {
        const __bf16* ap = w + (size_t)(o0 + lr) * CCH + k0 + hi * 8;
        v16bf af = cat16(*(const v8bf*)ap, *(const v8bf*)(ap + 16));
        #pragma unroll
        for (int t = 0; t < 4; ++t) {
            int n = n0 + t * 16 + lr;
            const __bf16* bp = ab + (size_t)n * CCH + k0 + hi * 8;
            v16bf bf_ = cat16(*(const v8bf*)bp, *(const v8bf*)(bp + 16));
            acc[t] = wmma_bf16(af, bf_, acc[t]);
        }
    }
    const float* xb = x   + (size_t)b * CCH * NPIX;
    float*       ob = out + (size_t)b * CCH * NPIX;
    #pragma unroll
    for (int t = 0; t < 4; ++t) {
        int n = n0 + t * 16 + lr;
        #pragma unroll
        for (int vv = 0; vv < 8; ++vv) {
            int o = o0 + vv + hi * 8;
            size_t idx = (size_t)o * NPIX + n;
            ob[idx] = xb[idx] + acc[t][vv] + bias[o];
        }
    }
}

// ---------------- launcher ----------------
extern "C" void kernel_launch(void* const* d_in, const int* in_sizes, int n_in,
                              void* d_out, int out_size, void* d_ws, size_t ws_size,
                              hipStream_t stream) {
    const float* x      = (const float*)d_in[0];
    const float* gn_w   = (const float*)d_in[1];
    const float* gn_b   = (const float*)d_in[2];
    const float* qkv_w  = (const float*)d_in[3];
    const float* qkv_b  = (const float*)d_in[4];
    const float* proj_w = (const float*)d_in[5];
    const float* proj_b = (const float*)d_in[6];
    float* out = (float*)d_out;

    char* ws = (char*)d_ws;
    __bf16* hT  = (__bf16*)(ws + HT_OFF);
    __bf16* q   = (__bf16*)(ws + Q_OFF);
    __bf16* k   = (__bf16*)(ws + K_OFF);
    __bf16* vT  = (__bf16*)(ws + V_OFF);
    __bf16* ao  = (__bf16*)(ws + AO_OFF);
    __bf16* qwb = (__bf16*)(ws + QW_OFF);
    __bf16* pwb = (__bf16*)(ws + PW_OFF);
    float*  mu  = (float*)(ws + MU_OFF);
    float*  rs  = (float*)(ws + RS_OFF);

    cvt_weights<<<3072, 256, 0, stream>>>(qkv_w, proj_w, qwb, pwb);
    gn_stats<<<BATCH * NG, 256, 0, stream>>>(x, mu, rs);
    gn_apply<<<(BATCH * CCH * NPIX) / 256, 256, 0, stream>>>(x, gn_w, gn_b, mu, rs, hT);
    qkv_gemm<<<6144, 128, 0, stream>>>(qwb, hT, qkv_b, q, k, vT);
    attn_flash<<<BATCH * (NPIX / 16), 128, 0, stream>>>(q, k, vT, ao);
    proj_gemm<<<2048, 128, 0, stream>>>(pwb, ao, proj_b, x, out);
}
